// GladiusAttention_17944373363113
// MI455X (gfx1250) — matrix operand order repaired and checked by reading, not compiled, not run on hardware
//
#include <hip/hip_runtime.h>
#include <cstdint>
#include <cstddef>

// Problem constants (match reference)
#define L_  2048
#define D_  2048
#define H_  32
#define KV_ 8
#define HD_ 64
#define HHD (H_*HD_)   // 2048
#define KVD (KV_*HD_)  // 512

typedef __attribute__((ext_vector_type(16))) __bf16 v16bf;
typedef __attribute__((ext_vector_type(8)))  float  v8f;
typedef __attribute__((ext_vector_type(4)))  unsigned int u32x4;
typedef __attribute__((ext_vector_type(8)))  int i32x8;
typedef __attribute__((ext_vector_type(4)))  int i32x4;

#define WMMA_BF16(a,b,c) \
  __builtin_amdgcn_wmma_f32_16x16x32_bf16(false,(a),false,(b),(short)0,(c),false,false)

// Tensor Data Mover availability (5-arg on ROCm7.2/clang-22, 6-arg on clang-23+)
#if defined(__has_builtin)
# if __has_builtin(__builtin_amdgcn_tensor_load_to_lds)
#  define HAVE_TDM 1
# endif
#endif
#ifndef HAVE_TDM
# define HAVE_TDM 0
#endif

// Surface the chosen data-movement path in compiler stderr (compile-only loop).
#if HAVE_TDM
# warning "CDNA5: TDM path active (tensor_load_to_lds + s_wait_tensorcnt)"
#else
# warning "CDNA5: tensor_load_to_lds builtin NOT available; cooperative-load fallback"
#endif

// ---------------------------------------------------------------------------
// Fragment loaders (contiguous k order chosen by this code, same for A and B):
// A (16x32 bf16): lanes 0-15 row M=lane, halves 0..15 = K0..15
//                 lanes 16-31 row M=lane-16, halves 0..15 = K16..31
// B (32x16 bf16): lane n%16 = column N, halves 0..15 = K0..15 (lanes<16)
//                 or K16..31 (lanes>=16)
// C/D (16x16 f32): VGPR r, lanes 0-15 -> M=r, lanes 16-31 -> M=r+8; N=lane%16
// ---------------------------------------------------------------------------
__device__ __forceinline__ v16bf frag_a(const __bf16* base, int ld, int lane) {
  const __bf16* p = base + (lane & 15) * ld;
  const int k0 = (lane < 16) ? 0 : 16;
  v16bf f;
#pragma unroll
  for (int i = 0; i < 16; ++i) f[i] = p[k0 + i];
  return f;
}

__device__ __forceinline__ v16bf frag_b(const __bf16* base, int ld, int lane) {
  const __bf16* p = base + (lane & 15) * ld + ((lane < 16) ? 0 : 16);
  v16bf f;
#pragma unroll
  for (int i = 0; i < 16; ++i) f[i] = p[i];
  return f;
}

#if HAVE_TDM
// ---------------------------------------------------------------------------
// TDM: async-DMA a 2D bf16 tile (tile_d1 rows x tile_d0 elems) from global
// into LDS, inserting 16B of padding after each 64B row so the LDS layout
// matches a [rows][40] __bf16 array (bank-conflict padding).
// Descriptor packing per CDNA5 ISA.
// ---------------------------------------------------------------------------
__device__ __forceinline__ void tdm_load_tile_bf16(
    unsigned lds_addr, const __bf16* gaddr,
    unsigned tensor_d0, unsigned tensor_d1,
    unsigned long long stride0_elems,
    unsigned tile_d0, unsigned tile_d1) {
  unsigned long long ga = (unsigned long long)(uintptr_t)gaddr;
  u32x4 g0;
  g0[0] = 1u;                                            // count=1, user D#
  g0[1] = lds_addr;                                      // lds_addr[31:0]
  g0[2] = (unsigned)(ga & 0xFFFFFFFFu);                  // global_addr[31:0]
  g0[3] = (unsigned)((ga >> 32) & 0x01FFFFFFu)           // global_addr[56:32]
        | (2u << 30);                                    // type = 2 ("image")
  i32x8 g1;
  g1[0] = (int)((1u << 16)        // data_size = 1 (2 bytes)
              | (1u << 20)        // pad_enable
              | (3u << 22)        // pad_interval: every 16 dwords (64B row)
              | (3u << 25));      // pad_amount: 4 dwords (16B)
  g1[1] = (int)((tensor_d0 & 0xFFFFu) << 16);            // tensor_dim0[15:0]
  g1[2] = (int)(((tensor_d0 >> 16) & 0xFFFFu)            // tensor_dim0[31:16]
              | ((tensor_d1 & 0xFFFFu) << 16));          // tensor_dim1[15:0]
  g1[3] = (int)(((tensor_d1 >> 16) & 0xFFFFu)            // tensor_dim1[31:16]
              | ((tile_d0 & 0xFFFFu) << 16));            // tile_dim0
  g1[4] = (int)(tile_d1 & 0xFFFFu);                      // tile_dim1 (tile_dim2=0)
  g1[5] = (int)(stride0_elems & 0xFFFFFFFFull);          // dim0_stride[31:0]
  g1[6] = (int)((stride0_elems >> 32) & 0xFFFFull);      // dim0_stride[47:32]
  g1[7] = 0;
  i32x4 z4 = {0, 0, 0, 0};                               // 2D tensor: groups 2/3 unused
#if __clang_major__ >= 23
  i32x8 z8 = {0, 0, 0, 0, 0, 0, 0, 0};
  __builtin_amdgcn_tensor_load_to_lds(g0, g1, z4, z4, z8, 0);
#else
  __builtin_amdgcn_tensor_load_to_lds(g0, g1, z4, z4, 0);
#endif
}

__device__ __forceinline__ unsigned lds_byte_addr(const void* p) {
  // Flat LDS addresses map to LDS via addr[31:0] (CDNA5 ISA)
  return (unsigned)(uintptr_t)p;
}
#endif  // HAVE_TDM

// ---------------------------------------------------------------------------
// fp32 -> bf16 conversion
// ---------------------------------------------------------------------------
__global__ void f32_to_bf16_kernel(const float* __restrict__ in,
                                   __bf16* __restrict__ out, int n) {
  int i = blockIdx.x * 256 + threadIdx.x;
  if (i < n) out[i] = (__bf16)in[i];
}

// ---------------------------------------------------------------------------
// bf16 WMMA GEMM: C[M,N] = A[M,K] * B[N,K]^T
// Block tile 128x128, K-step 32, double-buffered LDS fed by TDM.
// 8 waves, each wave computes a 32(M) x 64(N) tile = 8 v_wmma per K-step.
// ---------------------------------------------------------------------------
#define BM 128
#define BN 128
#define BK 32
#define LDP 40   // 32 halves + 16B pad per row (matches TDM pad descriptor)

template <typename OutT>
__global__ __launch_bounds__(256)
void gemm_bf16_wmma(const __bf16* __restrict__ A,  // M x K
                    const __bf16* __restrict__ B,  // N x K
                    OutT* __restrict__ C,          // M x N
                    int M, int N, int K) {
  __shared__ __bf16 sA[2][BM][LDP];
  __shared__ __bf16 sB[2][BN][LDP];

  const int tid  = threadIdx.x;
  const int lane = tid & 31;
  const int wave = tid >> 5;          // 0..7
  const int wm   = (wave >> 1) * 32;  // 0,32,64,96
  const int wn   = (wave & 1) * 64;   // 0,64

  const int m0 = blockIdx.y * BM;
  const int n0 = blockIdx.x * BN;

  v8f acc[2][4];
#pragma unroll
  for (int i = 0; i < 2; ++i)
#pragma unroll
    for (int j = 0; j < 4; ++j)
#pragma unroll
      for (int r = 0; r < 8; ++r) acc[i][j][r] = 0.0f;

  const int nk = K / BK;

#if HAVE_TDM
  // Prologue: DMA first K-tile of A and B into buffer 0 (one wave issues).
  if (wave == 0) {
    tdm_load_tile_bf16(lds_byte_addr(&sA[0][0][0]), A + (size_t)m0 * K,
                       (unsigned)K, (unsigned)M, (unsigned long long)K, BK, BM);
    tdm_load_tile_bf16(lds_byte_addr(&sB[0][0][0]), B + (size_t)n0 * K,
                       (unsigned)K, (unsigned)N, (unsigned long long)K, BK, BN);
  }
#endif

  for (int kb = 0; kb < nk; ++kb) {
    const int buf = kb & 1;

#if HAVE_TDM
    if (wave == 0) __builtin_amdgcn_s_wait_tensorcnt(0);
    __syncthreads();  // tile `buf` visible to all waves
    // Overlap: kick off DMA of the next K-tile into the other buffer while
    // this buffer is consumed by WMMA.
    if (wave == 0 && kb + 1 < nk) {
      const int kn = (kb + 1) * BK;
      tdm_load_tile_bf16(lds_byte_addr(&sA[buf ^ 1][0][0]),
                         A + (size_t)m0 * K + kn, (unsigned)K, (unsigned)M,
                         (unsigned long long)K, BK, BM);
      tdm_load_tile_bf16(lds_byte_addr(&sB[buf ^ 1][0][0]),
                         B + (size_t)n0 * K + kn, (unsigned)K, (unsigned)N,
                         (unsigned long long)K, BK, BN);
    }
#else
    // Fallback: cooperative staged load (no TDM builtin on this toolchain).
    {
      const int r = tid >> 1, c = (tid & 1) * 16;
      const __bf16* ga = A + (size_t)(m0 + r) * K + kb * BK + c;
      const __bf16* gb = B + (size_t)(n0 + r) * K + kb * BK + c;
#pragma unroll
      for (int i = 0; i < 16; ++i) sA[buf][r][c + i] = ga[i];
#pragma unroll
      for (int i = 0; i < 16; ++i) sB[buf][r][c + i] = gb[i];
    }
    __syncthreads();
#endif

    v16bf af[2], bfr[4];
#pragma unroll
    for (int i = 0; i < 2; ++i)
      af[i] = frag_a(&sA[buf][wm + 16 * i][0], LDP, lane);
#pragma unroll
    for (int j = 0; j < 4; ++j)
      bfr[j] = frag_b(&sB[buf][wn + 16 * j][0], LDP, lane);

#pragma unroll
    for (int i = 0; i < 2; ++i)
#pragma unroll
      for (int j = 0; j < 4; ++j)
        acc[i][j] = WMMA_BF16(af[i], bfr[j], acc[i][j]);

    __syncthreads();  // done reading `buf` before it is re-filled
  }

  // store C (f32 C/D fragment layout)
#pragma unroll
  for (int i = 0; i < 2; ++i)
#pragma unroll
    for (int j = 0; j < 4; ++j) {
      const int col = n0 + wn + 16 * j + (lane & 15);
      const int rb0 = m0 + wm + 16 * i + 8 * (lane >> 4);
#pragma unroll
      for (int r = 0; r < 8; ++r)
        C[(size_t)(rb0 + r) * N + col] = (OutT)acc[i][j][r];
    }
}

// ---------------------------------------------------------------------------
// In-place RoPE on bf16 [L][nheads*64], with optional scale (softmax 1/sqrt(HD)
// folded into Q). Thread per (l, head, i<32) rotation pair.
// ---------------------------------------------------------------------------
__global__ void rope_scale_kernel(__bf16* __restrict__ buf, int nheads,
                                  float scale, int total) {
  int idx = blockIdx.x * 256 + threadIdx.x;
  if (idx >= total) return;
  int i = idx & 31;
  int h = (idx >> 5) % nheads;
  int l = idx / (32 * nheads);
  __bf16* p = buf + ((size_t)l * nheads + h) * HD_;
  float x1 = (float)p[i], x2 = (float)p[i + 32];
  // inv_freq = theta^(-2i/64) = exp(-i * ln(10000)/32)
  float inv = __expf(-(float)i * 0.28782313662425575f);
  float ang = (float)l * inv;
  float c = __cosf(ang), s = __sinf(ang);
  p[i]      = (__bf16)((x1 * c - x2 * s) * scale);
  p[i + 32] = (__bf16)((x2 * c + x1 * s) * scale);
}

// ---------------------------------------------------------------------------
// Transpose V: [L][KV*HD] -> [KV*HD][L] so PV B-fragments load contiguously.
// ---------------------------------------------------------------------------
__global__ void transpose_v_kernel(const __bf16* __restrict__ Vb,
                                   __bf16* __restrict__ Vt) {
  int i = blockIdx.x * 256 + threadIdx.x;
  if (i >= L_ * KVD) return;
  int c = i % KVD, l = i / KVD;
  Vt[(size_t)c * L_ + l] = Vb[i];
}

// ---------------------------------------------------------------------------
// Flash attention: one wave per (head, 32-query tile), kv step 32.
// Two M-tiles per wave share every K/V fragment -> 2x WMMA per fragment load.
// Q pre-scaled by 1/sqrt(HD) and roped; K roped; Vt transposed.
// ---------------------------------------------------------------------------
__global__ __launch_bounds__(32)
void attn_kernel(const __bf16* __restrict__ Qb,   // [L][H*HD]
                 const __bf16* __restrict__ Kb,   // [L][KV*HD]
                 const __bf16* __restrict__ Vt,   // [KV*HD][L]
                 __bf16* __restrict__ Ob) {       // [L][H*HD]
  const int lane = threadIdx.x & 31;
  const int h  = blockIdx.y;
  const int kh = h >> 2;                 // G = 4
  const int q0 = blockIdx.x * 32;

  __shared__ __bf16 sP[32][40];

  // Q fragments: 2 M-tiles x 2 d-chunks (each 16x32 A-frag), from global
  v16bf qf[2][2];
#pragma unroll
  for (int mi = 0; mi < 2; ++mi) {
    const __bf16* qrow =
        Qb + (size_t)(q0 + 16 * mi + (lane & 15)) * HHD + h * HD_;
    const int k0 = (lane < 16) ? 0 : 16;
#pragma unroll
    for (int c = 0; c < 2; ++c) {
      const __bf16* p = qrow + 32 * c;
      v16bf f;
#pragma unroll
      for (int i = 0; i < 16; ++i) f[i] = p[k0 + i];
      qf[mi][c] = f;
    }
  }

  v8f oacc[2][4];
#pragma unroll
  for (int mi = 0; mi < 2; ++mi)
#pragma unroll
    for (int nt = 0; nt < 4; ++nt)
#pragma unroll
      for (int r = 0; r < 8; ++r) oacc[mi][nt][r] = 0.0f;
  float mrow[2][8], lrow[2][8];
#pragma unroll
  for (int mi = 0; mi < 2; ++mi)
#pragma unroll
    for (int r = 0; r < 8; ++r) { mrow[mi][r] = -1e30f; lrow[mi][r] = 0.0f; }

  const int nblocks = (q0 >> 5) + 1;  // causal: kv cols [0, q0+32)
  for (int jb = 0; jb < nblocks; ++jb) {
    const int j0 = jb * 32;

    // ---- S = Q K^T : 2 M-tiles x 2 N-tiles; K-frags shared across M-tiles
    v8f s[2][2];
#pragma unroll
    for (int mi = 0; mi < 2; ++mi)
#pragma unroll
      for (int t = 0; t < 2; ++t)
#pragma unroll
        for (int r = 0; r < 8; ++r) s[mi][t][r] = 0.0f;

#pragma unroll
    for (int t = 0; t < 2; ++t) {
      const __bf16* krow = Kb + (size_t)(j0 + 16 * t + (lane & 15)) * KVD
                              + kh * HD_ + ((lane < 16) ? 0 : 16);
#pragma unroll
      for (int c = 0; c < 2; ++c) {
        v16bf kf;
#pragma unroll
        for (int i = 0; i < 16; ++i) kf[i] = krow[32 * c + i];
#pragma unroll
        for (int mi = 0; mi < 2; ++mi)
          s[mi][t] = WMMA_BF16(qf[mi][c], kf, s[mi][t]);
      }
    }

    // ---- causal mask (per M-tile; only diagonal-adjacent blocks)
#pragma unroll
    for (int mi = 0; mi < 2; ++mi) {
      if (j0 + 31 > q0 + 16 * mi) {
        const int rowb = q0 + 16 * mi + 8 * (lane >> 4);
#pragma unroll
        for (int t = 0; t < 2; ++t) {
          const int col = j0 + 16 * t + (lane & 15);
#pragma unroll
          for (int r = 0; r < 8; ++r)
            if (col > rowb + r) s[mi][t][r] = -1e30f;
        }
      }
    }

    // ---- online softmax (row stats replicated across each 16-lane half)
#pragma unroll
    for (int mi = 0; mi < 2; ++mi) {
#pragma unroll
      for (int r = 0; r < 8; ++r) {
        float mx = fmaxf(s[mi][0][r], s[mi][1][r]);
#pragma unroll
        for (int off = 1; off < 16; off <<= 1)
          mx = fmaxf(mx, __shfl_xor(mx, off, 32));
        float mnew  = fmaxf(mrow[mi][r], mx);
        float alpha = __expf(mrow[mi][r] - mnew);
        float p0 = __expf(s[mi][0][r] - mnew);
        float p1 = __expf(s[mi][1][r] - mnew);
        float ps = p0 + p1;
#pragma unroll
        for (int off = 1; off < 16; off <<= 1)
          ps += __shfl_xor(ps, off, 32);
        lrow[mi][r] = lrow[mi][r] * alpha + ps;
        mrow[mi][r] = mnew;
        s[mi][0][r] = p0;
        s[mi][1][r] = p1;
#pragma unroll
        for (int nt = 0; nt < 4; ++nt) oacc[mi][nt][r] *= alpha;
      }
    }

    // ---- bounce P through LDS to repack C-layout -> A-fragment layout
#pragma unroll
    for (int mi = 0; mi < 2; ++mi)
#pragma unroll
      for (int t = 0; t < 2; ++t)
#pragma unroll
        for (int r = 0; r < 8; ++r)
          sP[16 * mi + r + 8 * (lane >> 4)][16 * t + (lane & 15)] =
              (__bf16)s[mi][t][r];
    __syncthreads();

    v16bf pf[2];
#pragma unroll
    for (int mi = 0; mi < 2; ++mi) {
      const __bf16* p = &sP[16 * mi + (lane & 15)][0];
      const int k0 = (lane < 16) ? 0 : 16;
      v16bf f;
#pragma unroll
      for (int i = 0; i < 16; ++i) f[i] = p[k0 + i];
      pf[mi] = f;
    }

    // ---- O += P * V  (K=32 kv, N=64); V-frags shared across M-tiles
#pragma unroll
    for (int nt = 0; nt < 4; ++nt) {
      const __bf16* vp = Vt + ((size_t)(kh * HD_ + nt * 16 + (lane & 15))) * L_
                            + j0 + ((lane < 16) ? 0 : 16);
      v16bf vf;
#pragma unroll
      for (int i = 0; i < 16; ++i) vf[i] = vp[i];
#pragma unroll
      for (int mi = 0; mi < 2; ++mi)
        oacc[mi][nt] = WMMA_BF16(pf[mi], vf, oacc[mi][nt]);
    }
    __syncthreads();
  }

  // ---- normalize and store
#pragma unroll
  for (int mi = 0; mi < 2; ++mi)
#pragma unroll
    for (int nt = 0; nt < 4; ++nt) {
      const int gcol = h * HD_ + nt * 16 + (lane & 15);
#pragma unroll
      for (int r = 0; r < 8; ++r) {
        const int grow = q0 + 16 * mi + r + 8 * (lane >> 4);
        Ob[(size_t)grow * HHD + gcol] = (__bf16)(oacc[mi][nt][r] / lrow[mi][r]);
      }
    }
}

// ---------------------------------------------------------------------------
// Host launcher
// ---------------------------------------------------------------------------
extern "C" void kernel_launch(void* const* d_in, const int* in_sizes, int n_in,
                              void* d_out, int out_size, void* d_ws, size_t ws_size,
                              hipStream_t stream) {
  (void)in_sizes; (void)n_in; (void)out_size; (void)ws_size;

  const float* x  = (const float*)d_in[0];
  const float* Wq = (const float*)d_in[1];
  const float* Wk = (const float*)d_in[2];
  const float* Wv = (const float*)d_in[3];
  const float* Wo = (const float*)d_in[4];
  float* out = (float*)d_out;

  char* ws = (char*)d_ws;
  size_t off = 0;
  auto alloc = [&](size_t bytes) -> char* {
    char* p = ws + off;
    off += (bytes + 255) & ~(size_t)255;
    return p;
  };
  __bf16* xb  = (__bf16*)alloc((size_t)L_ * D_ * 2);
  __bf16* Wqb = (__bf16*)alloc((size_t)HHD * D_ * 2);
  __bf16* Wkb = (__bf16*)alloc((size_t)KVD * D_ * 2);
  __bf16* Wvb = (__bf16*)alloc((size_t)KVD * D_ * 2);
  __bf16* Wob = (__bf16*)alloc((size_t)D_ * HHD * 2);
  __bf16* Qb  = (__bf16*)alloc((size_t)L_ * HHD * 2);
  __bf16* Kb  = (__bf16*)alloc((size_t)L_ * KVD * 2);
  __bf16* Vb  = (__bf16*)alloc((size_t)L_ * KVD * 2);
  __bf16* Vt  = (__bf16*)alloc((size_t)KVD * L_ * 2);
  __bf16* Ab  = (__bf16*)alloc((size_t)L_ * HHD * 2);

  int n;
  n = L_ * D_;
  f32_to_bf16_kernel<<<(n + 255) / 256, 256, 0, stream>>>(x, xb, n);
  n = HHD * D_;
  f32_to_bf16_kernel<<<(n + 255) / 256, 256, 0, stream>>>(Wq, Wqb, n);
  n = KVD * D_;
  f32_to_bf16_kernel<<<(n + 255) / 256, 256, 0, stream>>>(Wk, Wkb, n);
  f32_to_bf16_kernel<<<(n + 255) / 256, 256, 0, stream>>>(Wv, Wvb, n);
  n = D_ * HHD;
  f32_to_bf16_kernel<<<(n + 255) / 256, 256, 0, stream>>>(Wo, Wob, n);

  // QKV projections
  gemm_bf16_wmma<__bf16><<<dim3(HHD / BN, L_ / BM), 256, 0, stream>>>(
      xb, Wqb, Qb, L_, HHD, D_);
  gemm_bf16_wmma<__bf16><<<dim3(KVD / BN, L_ / BM), 256, 0, stream>>>(
      xb, Wkb, Kb, L_, KVD, D_);
  gemm_bf16_wmma<__bf16><<<dim3(KVD / BN, L_ / BM), 256, 0, stream>>>(
      xb, Wvb, Vb, L_, KVD, D_);

  // RoPE (scale folded into Q) + V transpose
  n = L_ * H_ * 32;
  rope_scale_kernel<<<(n + 255) / 256, 256, 0, stream>>>(Qb, H_, 0.125f, n);
  n = L_ * KV_ * 32;
  rope_scale_kernel<<<(n + 255) / 256, 256, 0, stream>>>(Kb, KV_, 1.0f, n);
  n = L_ * KVD;
  transpose_v_kernel<<<(n + 255) / 256, 256, 0, stream>>>(Vb, Vt);

  // Flash attention (32 queries per wave)
  attn_kernel<<<dim3(L_ / 32, H_), 32, 0, stream>>>(Qb, Kb, Vt, Ab);

  // Output projection (fp32 result)
  gemm_bf16_wmma<float><<<dim3(D_ / BN, L_ / BM), 256, 0, stream>>>(
      Ab, Wob, out, L_, D_, HHD);
}
